// HangulViT_2078764171912
// MI455X (gfx1250) — matrix operand order repaired and checked by reading, not compiled, not run on hardware
//
#include <hip/hip_runtime.h>
#include <cstdint>
#include <cstddef>

// ---------------------------------------------------------------------------
// Model constants
// ---------------------------------------------------------------------------
#define BB   64
#define SS   64
#define DD   512
#define HH   8
#define DHH  64
#define FFD  2048
#define LL   6
#define VV   54
#define PADT 53
#define NFF  121

typedef __attribute__((ext_vector_type(16))) _Float16 v16h;
typedef __attribute__((ext_vector_type(8)))  _Float16 v8h;
typedef __attribute__((ext_vector_type(4)))  _Float16 v4h;
typedef __attribute__((ext_vector_type(8)))  float    v8f;

#define WMMA_F16(a, b, c) \
  __builtin_amdgcn_wmma_f32_16x16x32_f16(false, (a), false, (b), (short)0, (c), false, false)

// ---------------------------------------------------------------------------
// WMMA fragment loaders from LDS (layouts per CDNA5 ISA 7.12.2)
// A 16x32 f16: lane l holds row M=l%16; half h -> K = (h<8 ? h : h+8) + (l/16)*8
// B 32x16 f16: lane l holds col N=l%16; half h -> K = (h<8 ? h : h+8) + (l/16)*8
// C/D 16x16 f32: vgpr r, lane l -> M = r + 8*(l/16), N = l%16
// ---------------------------------------------------------------------------
__device__ __forceinline__ v16h frag_a(const _Float16* p, int laneHi) {
  v8h lo = *(const v8h*)(p + laneHi * 8);
  v8h hi = *(const v8h*)(p + 16 + laneHi * 8);
  v16h a;
#pragma unroll
  for (int i = 0; i < 8; ++i) { a[i] = lo[i]; a[i + 8] = hi[i]; }
  return a;
}

__device__ __forceinline__ v16h frag_b(const _Float16* p, int laneHi) {
  v8h lo = *(const v8h*)(p + laneHi * 8);
  v8h hi = *(const v8h*)(p + 16 + laneHi * 8);
  v16h b;
#pragma unroll
  for (int i = 0; i < 8; ++i) { b[i] = lo[i]; b[i + 8] = hi[i]; }
  return b;
}

// ---------------------------------------------------------------------------
// Weight convert + transpose:  W[K,N] f32  ->  Wt[N,K] f16
// ---------------------------------------------------------------------------
__global__ __launch_bounds__(256) void wcvt_kernel(
    const float* __restrict__ W, _Float16* __restrict__ Wt, int K, int N)
{
  __shared__ _Float16 tile[32][33];
  const int k0 = blockIdx.x * 32;
  const int n0 = blockIdx.y * 32;
  const int tx = threadIdx.x;          // 0..31
  const int ty = threadIdx.y;          // 0..7
#pragma unroll
  for (int p = 0; p < 4; ++p) {
    const int k = k0 + ty + p * 8;
    const int n = n0 + tx;
    float v = (k < K && n < N) ? W[(size_t)k * N + n] : 0.f;
    tile[tx][ty + p * 8] = (_Float16)v;     // tile[n][k]
  }
  __syncthreads();
#pragma unroll
  for (int p = 0; p < 4; ++p) {
    const int n = n0 + ty + p * 8;
    const int k = k0 + tx;
    if (n < N && k < K) Wt[(size_t)n * K + k] = tile[ty + p * 8][tx];
  }
}

// ---------------------------------------------------------------------------
// Tiled WMMA GEMM:  C[M,N] = A[M,K] @ W[K,N] + bias[N]  (optional ReLU)
// A is f32 [M,K]; weights come pre-converted/transposed: Wt f16 [N,K].
// Block tile 128x64, 8 waves (all in M); each wave: 4 WMMAs per 32-wide K step.
// Weight tiles move via GLOBAL_LOAD_ASYNC_TO_LDS_B128 (ASYNCcnt pipeline) when
// N is a multiple of 64; A tiles are f32->f16 converted through registers.
// ---------------------------------------------------------------------------
#define ATS 40  // padded LDS row stride (halves), 16B aligned

struct ARegs { v4h a[4]; };

__device__ __forceinline__ ARegs load_a(
    const float* __restrict__ A, int M, int K, int mbase, int kb, int tid, bool guard)
{
  ARegs r;
#pragma unroll
  for (int p = 0; p < 4; ++p) {
    const int task = tid + 256 * p;       // 0..1023
    const int row  = task >> 3;           // 0..127
    const int cg   = task & 7;            // 0..7 (x4 floats)
    const int m    = mbase + row;
    float4 f = make_float4(0.f, 0.f, 0.f, 0.f);
    if (!guard || m < M) f = *(const float4*)&A[(size_t)m * K + kb + cg * 4];
    v4h h;
    h[0] = (_Float16)f.x; h[1] = (_Float16)f.y;
    h[2] = (_Float16)f.z; h[3] = (_Float16)f.w;
    r.a[p] = h;
  }
  return r;
}

__device__ __forceinline__ void store_a(const ARegs& r, _Float16* Ah, int tid)
{
#pragma unroll
  for (int p = 0; p < 4; ++p) {
    const int task = tid + 256 * p;
    const int row  = task >> 3;
    const int cg   = task & 7;
    *(v4h*)&Ah[row * ATS + cg * 4] = r.a[p];
  }
}

__device__ __forceinline__ v8h load_w(
    const _Float16* __restrict__ Wt, int N, int K, int nbase, int kb, int tid)
{
  const int n  = tid >> 2;
  const int kg = tid & 3;
  v8h w;
#pragma unroll
  for (int i = 0; i < 8; ++i) w[i] = (_Float16)0.f;
  if (nbase + n < N) w = *(const v8h*)&Wt[(size_t)(nbase + n) * K + kb + kg * 8];
  return w;
}

__device__ __forceinline__ void store_w(v8h w, _Float16* Wh, int tid)
{
  const int n  = tid >> 2;
  const int kg = tid & 3;
  *(v8h*)&Wh[n * ATS + kg * 8] = w;
}

// Async 16B copy global f16 -> LDS (per-lane LDS dest keeps the padded stride).
// HIP __shared__ generic pointers carry the LDS offset in the low 32 bits.
__device__ __forceinline__ void async_w(
    const _Float16* __restrict__ Wt, _Float16* Wh, int K, int nbase, int kb, int tid)
{
  const int n  = tid >> 2;
  const int kg = tid & 3;
  const _Float16* src = &Wt[(size_t)(nbase + n) * K + kb + kg * 8];
  _Float16* dst = &Wh[n * ATS + kg * 8];
  const unsigned lds = (unsigned)(uintptr_t)dst;
  asm volatile("global_load_async_to_lds_b128 %0, %1, off"
               :: "v"(lds), "v"((unsigned long long)(uintptr_t)src)
               : "memory");
}

__device__ __forceinline__ void compute_tile(
    const _Float16* Ah, const _Float16* Wh, v8f acc[4],
    int wv, int laneLo, int laneHi)
{
  const v16h af = frag_a(&Ah[(16 * wv + laneLo) * ATS], laneHi);
#pragma unroll
  for (int t = 0; t < 4; ++t) {
    const v16h bf = frag_b(&Wh[(16 * t + laneLo) * ATS], laneHi);
    acc[t] = WMMA_F16(af, bf, acc[t]);
  }
}

template<bool ASYNCW, bool MGUARD>
__global__ __launch_bounds__(256) void gemm_bias_kernel(
    const float* __restrict__ A, const _Float16* __restrict__ Wt,
    const float* __restrict__ bias, float* __restrict__ C,
    int M, int N, int K, int relu)
{
  __shared__ __align__(16) _Float16 Ah[2][128 * ATS];
  __shared__ __align__(16) _Float16 Wh[2][64 * ATS];

  const int tid    = threadIdx.x;
  const int lane   = tid & 31;
  const int wv     = tid >> 5;            // 0..7 -> rows 16*wv
  const int laneLo = lane & 15;
  const int laneHi = lane >> 4;
  const int mbase  = blockIdx.y * 128;
  const int nbase  = blockIdx.x * 64;

  v8f acc[4];
#pragma unroll
  for (int t = 0; t < 4; ++t)
#pragma unroll
    for (int i = 0; i < 8; ++i) acc[t][i] = 0.f;

  ARegs ar = load_a(A, M, K, mbase, 0, tid, MGUARD);
  if (ASYNCW) {
    async_w(Wt, &Wh[0][0], K, nbase, 0, tid);
  } else {
    v8h w = load_w(Wt, N, K, nbase, 0, tid);
    store_w(w, &Wh[0][0], tid);
  }
  store_a(ar, &Ah[0][0], tid);
  int cur = 0;

  if (ASYNCW) {
    for (int kb = 32; kb < K; kb += 32) {
      ar = load_a(A, M, K, mbase, kb, tid, MGUARD);
      if (kb + 32 < K) {
        int pm = mbase + (tid >> 3); if (MGUARD && pm >= M) pm = M - 1;
        __builtin_prefetch(&A[(size_t)pm * K + kb + 32], 0, 1);
      }
      __syncthreads();                                    // next buf free of readers
      async_w(Wt, &Wh[cur ^ 1][0], K, nbase, kb, tid);    // ASYNCcnt++
      asm volatile("s_wait_asynccnt 0x1" ::: "memory");   // older async (cur) done
      __syncthreads();                                    // all waves' cur complete
      compute_tile(&Ah[cur][0], &Wh[cur][0], acc, wv, laneLo, laneHi);
      store_a(ar, &Ah[cur ^ 1][0], tid);
      cur ^= 1;
    }
    asm volatile("s_wait_asynccnt 0x0" ::: "memory");
    __syncthreads();
    compute_tile(&Ah[cur][0], &Wh[cur][0], acc, wv, laneLo, laneHi);
  } else {
    for (int kb = 32; kb < K; kb += 32) {
      ar = load_a(A, M, K, mbase, kb, tid, MGUARD);
      v8h w = load_w(Wt, N, K, nbase, kb, tid);
      __syncthreads();
      compute_tile(&Ah[cur][0], &Wh[cur][0], acc, wv, laneLo, laneHi);
      store_a(ar, &Ah[cur ^ 1][0], tid);
      store_w(w, &Wh[cur ^ 1][0], tid);
      cur ^= 1;
    }
    __syncthreads();
    compute_tile(&Ah[cur][0], &Wh[cur][0], acc, wv, laneLo, laneHi);
  }

#pragma unroll
  for (int t = 0; t < 4; ++t) {
    const int n = nbase + 16 * t + laneLo;
    if (n < N) {
      const float bv = bias[n];
#pragma unroll
      for (int r8 = 0; r8 < 8; ++r8) {
        const int m = mbase + 16 * wv + r8 + 8 * laneHi;
        if (!MGUARD || m < M) {
          float o = acc[t][r8] + bv;
          if (relu) o = fmaxf(o, 0.f);
          C[(size_t)m * N + n] = o;
        }
      }
    }
  }
}

// ---------------------------------------------------------------------------
// Fused attention per (b,h): scores = (q/sqrt(dh)) @ k^T -> mask -> softmax
// -> out = P @ v.  Sq = 64, dh = 64, Sk padded to 128.  All-WMMA.
// ---------------------------------------------------------------------------
#define SKP 128

__global__ __launch_bounds__(256) void attn_kernel(
    const float* __restrict__ q, const float* __restrict__ kbuf,
    const float* __restrict__ vbuf, const int* __restrict__ valid_row,
    const int* __restrict__ key_valid, float* __restrict__ o,
    int Sk, int is_self)
{
  __shared__ __align__(16) unsigned char smem[57344];
  _Float16* qh = (_Float16*)smem;                // [64][64]   (phase 1-2)
  _Float16* kh = (_Float16*)(smem + 8192);       // [128][64]  key-major (phase 1-2)
  float*    sc = (float*)(smem + 24576);         // [64][128]  scores (phase 2-3)
  _Float16* vt = (_Float16*)smem;                // [64][128]  v^T [dd][kk] (phase 4)
  _Float16* ah = (_Float16*)(smem + 24576);      // [64][128]  probs f16 (phase 4)

  const int tid    = threadIdx.x;
  const int lane   = tid & 31;
  const int wv     = tid >> 5;
  const int mw     = wv & 3;
  const int ng     = wv >> 2;
  const int laneLo = lane & 15;
  const int laneHi = lane >> 4;
  const int b      = blockIdx.x / HH;
  const int h      = blockIdx.x % HH;

  const size_t qoff = (size_t)b * SS * DD + (size_t)h * DHH;
  const size_t koff = (size_t)b * Sk * DD + (size_t)h * DHH;

  // Phase 1: stage q (pre-scaled by 1/sqrt(dh)) and k (key-major, zero-padded)
#pragma unroll
  for (int p = 0; p < 4; ++p) {
    const int idx = tid + 256 * p;             // 0..1023 float4 tasks
    const int s = idx >> 4, dd4 = idx & 15;
    const float4 f = *(const float4*)&q[qoff + (size_t)s * DD + dd4 * 4];
    v4h hv;
    hv[0] = (_Float16)(f.x * 0.125f); hv[1] = (_Float16)(f.y * 0.125f);
    hv[2] = (_Float16)(f.z * 0.125f); hv[3] = (_Float16)(f.w * 0.125f);
    *(v4h*)&qh[s * 64 + dd4 * 4] = hv;
  }
#pragma unroll
  for (int p = 0; p < 8; ++p) {
    const int idx = tid + 256 * p;             // 0..2047 float4 tasks
    const int n = idx >> 4, dd4 = idx & 15;
    float4 f = make_float4(0.f, 0.f, 0.f, 0.f);
    if (n < Sk) f = *(const float4*)&kbuf[koff + (size_t)n * DD + dd4 * 4];
    v4h hv;
    hv[0] = (_Float16)f.x; hv[1] = (_Float16)f.y;
    hv[2] = (_Float16)f.z; hv[3] = (_Float16)f.w;
    *(v4h*)&kh[n * 64 + dd4 * 4] = hv;
  }
  __syncthreads();

  // Phase 2: scores[64][128] = qh @ kh^T via WMMA (each wave: 16 rows x 64 cols)
  {
    v8f acc[4];
#pragma unroll
    for (int t = 0; t < 4; ++t)
#pragma unroll
      for (int i = 0; i < 8; ++i) acc[t][i] = 0.f;
    const _Float16* arow = &qh[(16 * mw + laneLo) * 64];
#pragma unroll
    for (int kb = 0; kb < 64; kb += 32) {
      const v16h af = frag_a(arow + kb, laneHi);
#pragma unroll
      for (int t = 0; t < 4; ++t) {
        const int col = 64 * ng + 16 * t + laneLo;     // key index
        const v16h bf = frag_b(&kh[col * 64] + kb, laneHi);
        acc[t] = WMMA_F16(af, bf, acc[t]);
      }
    }
#pragma unroll
    for (int t = 0; t < 4; ++t)
#pragma unroll
      for (int r = 0; r < 8; ++r)
        sc[(16 * mw + r + 8 * laneHi) * SKP + 64 * ng + 16 * t + laneLo] = acc[t][r];
  }
  __syncthreads();

  // Phase 3: masked softmax, one row per thread (threads 0..63)
  if (tid < 64) {
    const int sq = tid;
    const int vq = valid_row[b * SS + sq];
    float mx = -3.0e38f;
    for (int kk = 0; kk < Sk; ++kk) {
      float s = sc[sq * SKP + kk];
      int allowed = vq && (!is_self || (key_valid[b * SS + kk] && kk <= sq));
      s = allowed ? s : -1.0e9f;
      mx = fmaxf(mx, s);
    }
    float sum = 0.f;
    for (int kk = 0; kk < Sk; ++kk) {
      float s = sc[sq * SKP + kk];
      int allowed = vq && (!is_self || (key_valid[b * SS + kk] && kk <= sq));
      s = allowed ? s : -1.0e9f;
      float e = __expf(s - mx);
      sum += e;
      sc[sq * SKP + kk] = e;
    }
    const float inv = 1.f / sum;
    for (int kk = 0; kk < Sk; ++kk)  sc[sq * SKP + kk] *= inv;
    for (int kk = Sk; kk < SKP; ++kk) sc[sq * SKP + kk] = 0.f;
  }
  __syncthreads();

  // Phase 3b: f32->f16 probs in place (register-staged), stage v^T (zero-padded)
  {
    float tmp[32];
#pragma unroll
    for (int j = 0; j < 32; ++j) tmp[j] = sc[tid + 256 * j];
    __syncthreads();
#pragma unroll
    for (int j = 0; j < 32; ++j) ah[tid + 256 * j] = (_Float16)tmp[j];
#pragma unroll
    for (int p = 0; p < 8; ++p) {
      const int idx = tid + 256 * p;           // 0..2047 float4 tasks
      const int kk = idx >> 4, dd4 = idx & 15;
      float4 f = make_float4(0.f, 0.f, 0.f, 0.f);
      if (kk < Sk) f = *(const float4*)&vbuf[koff + (size_t)kk * DD + dd4 * 4];
      vt[(dd4 * 4 + 0) * SKP + kk] = (_Float16)f.x;
      vt[(dd4 * 4 + 1) * SKP + kk] = (_Float16)f.y;
      vt[(dd4 * 4 + 2) * SKP + kk] = (_Float16)f.z;
      vt[(dd4 * 4 + 3) * SKP + kk] = (_Float16)f.w;
    }
  }
  __syncthreads();

  // Phase 4: out[64][64] = P @ V via WMMA (each wave: 16 rows x 32 cols)
  {
    v8f o0, o1;
#pragma unroll
    for (int i = 0; i < 8; ++i) { o0[i] = 0.f; o1[i] = 0.f; }
    const _Float16* arow = &ah[(16 * mw + laneLo) * SKP];
#pragma unroll
    for (int kb = 0; kb < SKP; kb += 32) {
      const v16h af = frag_a(arow + kb, laneHi);
      const v16h b0 = frag_b(&vt[(32 * ng + laneLo) * SKP] + kb, laneHi);
      o0 = WMMA_F16(af, b0, o0);
      const v16h b1 = frag_b(&vt[(32 * ng + 16 + laneLo) * SKP] + kb, laneHi);
      o1 = WMMA_F16(af, b1, o1);
    }
    const size_t obase = (size_t)b * SS * DD + (size_t)h * DHH;
    const int dd0 = 32 * ng + laneLo;
#pragma unroll
    for (int r = 0; r < 8; ++r) {
      const int m = 16 * mw + r + 8 * laneHi;
      o[obase + (size_t)m * DD + dd0]      = o0[r];
      o[obase + (size_t)m * DD + dd0 + 16] = o1[r];
    }
  }
}

// ---------------------------------------------------------------------------
// Direct conv (stride 2, SAME: pad_begin = 0, pad_end = 1) + ReLU.
// transposed=1 writes out[b][oh*ow][co] (used for the final enc layout).
// ---------------------------------------------------------------------------
__global__ __launch_bounds__(256) void conv_kernel(
    const float* __restrict__ in, const float* __restrict__ w,
    const float* __restrict__ bias, float* __restrict__ out,
    int Cin, int Cout, int Hin, int Win, int Hout, int Wout,
    int transposed, int total)
{
  const int idx = blockIdx.x * 256 + threadIdx.x;
  if (idx >= total) return;
  const int ow = idx % Wout;
  int t = idx / Wout;
  const int oh = t % Hout; t /= Hout;
  const int co = t % Cout;
  const int b  = t / Cout;

  float acc = bias[co];
  const float* wb = w + (size_t)co * Cin * 9;
  for (int ci = 0; ci < Cin; ++ci) {
    const float* ip = in + ((size_t)b * Cin + ci) * Hin * Win;
    const float* wp = wb + ci * 9;
#pragma unroll
    for (int kh = 0; kh < 3; ++kh) {
      const int ih = 2 * oh + kh;
      if (ih >= Hin) continue;
#pragma unroll
      for (int kw = 0; kw < 3; ++kw) {
        const int iw = 2 * ow + kw;
        if (iw >= Win) continue;
        acc += ip[(size_t)ih * Win + iw] * wp[kh * 3 + kw];
      }
    }
  }
  acc = fmaxf(acc, 0.f);
  if (transposed)
    out[((size_t)b * Hout * Wout + (size_t)oh * Wout + ow) * Cout + co] = acc;
  else
    out[(((size_t)b * Cout + co) * Hout + oh) * Wout + ow] = acc;
}

// ---------------------------------------------------------------------------
// Embedding lookup * sqrt(D)
// ---------------------------------------------------------------------------
__global__ __launch_bounds__(256) void embed_kernel(
    const int* __restrict__ label, const float* __restrict__ emb,
    float* __restrict__ x)
{
  const int idx = blockIdx.x * 256 + threadIdx.x;   // B*S*D
  const int d = idx & (DD - 1);
  const int t = idx >> 9;
  const int lab = label[t];
  x[idx] = emb[(size_t)lab * DD + d] * 22.627416997969522f;  // sqrt(512)
}

// ---------------------------------------------------------------------------
// Mask precompute: valid_row (no pad in [0..s]), key_valid (label != PAD)
// ---------------------------------------------------------------------------
__global__ void mask_kernel(const int* __restrict__ label,
                            int* __restrict__ valid_row,
                            int* __restrict__ key_valid)
{
  const int b = blockIdx.x;
  if (threadIdx.x == 0) {
    int seen = 0;
    for (int s = 0; s < SS; ++s) {
      const int isp = (label[b * SS + s] == PADT);
      key_valid[b * SS + s] = !isp;
      seen += isp;
      valid_row[b * SS + s] = (seen == 0);
    }
  }
}

// ---------------------------------------------------------------------------
// x = LayerNorm(x + y) * g + b   (in place over x, one row of 512 per block)
// ---------------------------------------------------------------------------
__global__ __launch_bounds__(256) void add_ln_kernel(
    float* __restrict__ x, const float* __restrict__ y,
    const float* __restrict__ g, const float* __restrict__ bta)
{
  __shared__ float red[256];
  const int row = blockIdx.x;
  const int tid = threadIdx.x;
  const size_t base = (size_t)row * DD;
  const float a0 = x[base + tid] + y[base + tid];
  const float a1 = x[base + tid + 256] + y[base + tid + 256];
  red[tid] = a0 + a1;
  __syncthreads();
  for (int s = 128; s > 0; s >>= 1) {
    if (tid < s) red[tid] += red[tid + s];
    __syncthreads();
  }
  const float mean = red[0] * (1.f / DD);
  __syncthreads();
  const float d0 = a0 - mean, d1 = a1 - mean;
  red[tid] = d0 * d0 + d1 * d1;
  __syncthreads();
  for (int s = 128; s > 0; s >>= 1) {
    if (tid < s) red[tid] += red[tid + s];
    __syncthreads();
  }
  const float inv = rsqrtf(red[0] * (1.f / DD) + 1e-5f);
  x[base + tid]       = d0 * inv * g[tid] + bta[tid];
  x[base + tid + 256] = d1 * inv * g[tid + 256] + bta[tid + 256];
}

// ---------------------------------------------------------------------------
// log_softmax over V=54 (one row per thread)
// ---------------------------------------------------------------------------
__global__ __launch_bounds__(256) void lsm_kernel(
    const float* __restrict__ logits, float* __restrict__ out)
{
  const int row = blockIdx.x * 256 + threadIdx.x;
  if (row >= BB * SS) return;
  const float* r = logits + (size_t)row * VV;
  float mx = -3.0e38f;
  for (int j = 0; j < VV; ++j) mx = fmaxf(mx, r[j]);
  float sum = 0.f;
  for (int j = 0; j < VV; ++j) sum += __expf(r[j] - mx);
  const float lse = mx + __logf(sum);
  float* orow = out + (size_t)row * VV;
  for (int j = 0; j < VV; ++j) orow[j] = r[j] - lse;
}

// ---------------------------------------------------------------------------
// Host orchestration
// ---------------------------------------------------------------------------
extern "C" void kernel_launch(void* const* d_in, const int* in_sizes, int n_in,
                              void* d_out, int out_size, void* d_ws, size_t ws_size,
                              hipStream_t stream)
{
  (void)in_sizes; (void)n_in; (void)out_size; (void)ws_size;

  const float* image = (const float*)d_in[0];
  const int*   label = (const int*)d_in[1];
  const float* emb   = (const float*)d_in[2];
  const float* cw[4] = {(const float*)d_in[3], (const float*)d_in[5],
                        (const float*)d_in[7], (const float*)d_in[9]};
  const float* cb[4] = {(const float*)d_in[4], (const float*)d_in[6],
                        (const float*)d_in[8], (const float*)d_in[10]};
  const float* Wqkv_s = (const float*)d_in[11];
  const float* bqkv_s = (const float*)d_in[12];
  const float* Wo_s   = (const float*)d_in[13];
  const float* bo_s   = (const float*)d_in[14];
  const float* Wqkv_c = (const float*)d_in[15];
  const float* bqkv_c = (const float*)d_in[16];
  const float* Wo_c   = (const float*)d_in[17];
  const float* bo_c   = (const float*)d_in[18];
  const float* W1     = (const float*)d_in[19];
  const float* b1     = (const float*)d_in[20];
  const float* W2     = (const float*)d_in[21];
  const float* b2     = (const float*)d_in[22];
  const float* lng    = (const float*)d_in[23];
  const float* lnb    = (const float*)d_in[24];
  const float* Wg     = (const float*)d_in[25];
  const float* bg     = (const float*)d_in[26];
  float* out = (float*)d_out;

  // Workspace layout (float offsets)
  float* ws   = (float*)d_ws;
  float* bufA = ws;                                // 31,719,424 (conv1/conv3 out)
  float* bufB = ws + 33554432;                     // 15,859,712 (conv2 out, then x)
  float* enc  = ws + 50331648;                     // 3,964,928
  int*   valid_row = (int*)(ws + 54525952);        // 4096
  int*   key_valid = valid_row + BB * SS;          // 4096
  float* x    = bufB;
  float* qb   = bufA;                              // 2,097,152
  float* kb   = bufA + 2097152;                    // up to 3,964,928
  float* vb   = bufA + 6291456;                    // up to 3,964,928
  float* attno= bufA + 10485760;                   // 2,097,152
  float* t0   = bufA + 12582912;                   // 2,097,152
  float* ffb  = bufA + 14680064;                   // 8,388,608
  float* logits = t0;

  // f16 transposed weights region (halves), after the masks
  _Float16* w16 = (_Float16*)(ws + 54534144);
  const size_t LYR = 4194304;                      // halves per layer
  auto wt_slot = [&](int l, int s) -> _Float16* {  // s: 0..7 = qs,ks,vs,os,qc,kc,vc,oc
    return w16 + (size_t)l * LYR + (size_t)s * 262144;
  };
  auto w1_slot = [&](int l) -> _Float16* { return w16 + (size_t)l * LYR + 2097152; };
  auto w2_slot = [&](int l) -> _Float16* { return w16 + (size_t)l * LYR + 3145728; };
  _Float16* wg_t = w16 + 6 * LYR;                  // [54][512]

  auto wcvt = [&](const float* W, _Float16* Wt, int K, int N) {
    dim3 g((K + 31) / 32, (N + 31) / 32);
    wcvt_kernel<<<g, dim3(32, 8), 0, stream>>>(W, Wt, K, N);
  };
  auto gemm = [&](const float* A, const _Float16* Wt, const float* bias, float* C,
                  int M, int N, int K, int relu) {
    dim3 g((N + 63) / 64, (M + 127) / 128);
    const bool asy = (N % 64) == 0;
    const bool mg  = (M % 128) != 0;
    if (asy) {
      if (mg) gemm_bias_kernel<true,  true ><<<g, 256, 0, stream>>>(A, Wt, bias, C, M, N, K, relu);
      else    gemm_bias_kernel<true,  false><<<g, 256, 0, stream>>>(A, Wt, bias, C, M, N, K, relu);
    } else {
      if (mg) gemm_bias_kernel<false, true ><<<g, 256, 0, stream>>>(A, Wt, bias, C, M, N, K, relu);
      else    gemm_bias_kernel<false, false><<<g, 256, 0, stream>>>(A, Wt, bias, C, M, N, K, relu);
    }
  };

  // Masks
  mask_kernel<<<BB, 32, 0, stream>>>(label, valid_row, key_valid);

  // Weight pre-convert (f32 [K,N] -> f16 [N,K])
  for (int l = 0; l < LL; ++l) {
    for (int j = 0; j < 3; ++j) {
      wcvt(Wqkv_s + ((size_t)l * 3 + j) * DD * DD, wt_slot(l, j),     DD, DD);
      wcvt(Wqkv_c + ((size_t)l * 3 + j) * DD * DD, wt_slot(l, 4 + j), DD, DD);
    }
    wcvt(Wo_s + (size_t)l * DD * DD,  wt_slot(l, 3), DD, DD);
    wcvt(Wo_c + (size_t)l * DD * DD,  wt_slot(l, 7), DD, DD);
    wcvt(W1 + (size_t)l * DD * FFD,   w1_slot(l), DD, FFD);
    wcvt(W2 + (size_t)l * FFD * DD,   w2_slot(l), FFD, DD);
  }
  wcvt(Wg, wg_t, DD, VV);

  // CNN encoder: 176 -> 88 -> 44 -> 22 -> 11
  {
    const int Cin[4]  = {3, 64, 128, 256};
    const int Cout[4] = {64, 128, 256, 512};
    const int Hin[4]  = {176, 88, 44, 22};
    const int Hout[4] = {88, 44, 22, 11};
    const float* src[4] = {image, bufA, bufB, bufA};
    float* dst[4]       = {bufA, bufB, bufA, enc};
    for (int l = 0; l < 4; ++l) {
      const int total = BB * Cout[l] * Hout[l] * Hout[l];
      conv_kernel<<<(total + 255) / 256, 256, 0, stream>>>(
          src[l], cw[l], cb[l], dst[l], Cin[l], Cout[l],
          Hin[l], Hin[l], Hout[l], Hout[l], (l == 3) ? 1 : 0, total);
    }
  }

  // Token embedding
  embed_kernel<<<(BB * SS * DD) / 256, 256, 0, stream>>>(label, emb, x);

  // Decoder layers
  for (int i = 0; i < LL; ++i) {
    // --- self attention ---
    gemm(x, wt_slot(i, 0), bqkv_s + ((size_t)i * 3 + 0) * DD, qb, BB * SS, DD, DD, 0);
    gemm(x, wt_slot(i, 1), bqkv_s + ((size_t)i * 3 + 1) * DD, kb, BB * SS, DD, DD, 0);
    gemm(x, wt_slot(i, 2), bqkv_s + ((size_t)i * 3 + 2) * DD, vb, BB * SS, DD, DD, 0);
    attn_kernel<<<BB * HH, 256, 0, stream>>>(qb, kb, vb, valid_row, key_valid, attno, SS, 1);
    gemm(attno, wt_slot(i, 3), bo_s + (size_t)i * DD, t0, BB * SS, DD, DD, 0);
    add_ln_kernel<<<BB * SS, 256, 0, stream>>>(x, t0, lng + ((size_t)i * 3 + 0) * DD, lnb + ((size_t)i * 3 + 0) * DD);
    // --- cross attention ---
    gemm(x,   wt_slot(i, 4), bqkv_c + ((size_t)i * 3 + 0) * DD, qb, BB * SS,  DD, DD, 0);
    gemm(enc, wt_slot(i, 5), bqkv_c + ((size_t)i * 3 + 1) * DD, kb, BB * NFF, DD, DD, 0);
    gemm(enc, wt_slot(i, 6), bqkv_c + ((size_t)i * 3 + 2) * DD, vb, BB * NFF, DD, DD, 0);
    attn_kernel<<<BB * HH, 256, 0, stream>>>(qb, kb, vb, valid_row, key_valid, attno, NFF, 0);
    gemm(attno, wt_slot(i, 7), bo_c + (size_t)i * DD, t0, BB * SS, DD, DD, 0);
    add_ln_kernel<<<BB * SS, 256, 0, stream>>>(x, t0, lng + ((size_t)i * 3 + 1) * DD, lnb + ((size_t)i * 3 + 1) * DD);
    // --- FFN ---
    gemm(x,   w1_slot(i), b1 + (size_t)i * FFD, ffb, BB * SS, FFD, DD, 1);
    gemm(ffb, w2_slot(i), b2 + (size_t)i * DD,  t0,  BB * SS, DD, FFD, 0);
    add_ln_kernel<<<BB * SS, 256, 0, stream>>>(x, t0, lng + ((size_t)i * 3 + 2) * DD, lnb + ((size_t)i * 3 + 2) * DD);
  }

  // Vocab head + log_softmax -> out[0 .. B*S*V)
  gemm(x, wg_t, bg, logits, BB * SS, VV, DD, 0);
  lsm_kernel<<<(BB * SS + 255) / 256, 256, 0, stream>>>(logits, out);

  // Second tuple element: final x -> out[B*S*V ..)
  hipMemcpyAsync(out + (size_t)BB * SS * VV, x, (size_t)BB * SS * DD * sizeof(float),
                 hipMemcpyDeviceToDevice, stream);
}
